// CausalSelfAttention_34815004901518
// MI455X (gfx1250) — compile-verified, matrix operand order checked by reading, not run on hardware
//
#include <hip/hip_runtime.h>
#include <hip/hip_bf16.h>
#include <stdint.h>

typedef __bf16 bf16;
typedef __attribute__((ext_vector_type(16))) __bf16 v16bf;
typedef __attribute__((ext_vector_type(8)))  float  v8f;

#define B_   2
#define S_   2048
#define HID_ 2048
#define H_   16
#define HKV_ 4
#define D_   128

union FragU { v16bf v; uint4 q[2]; };

__device__ __forceinline__ v8f zero8() {
  v8f z = {0.f,0.f,0.f,0.f,0.f,0.f,0.f,0.f};
  return z;
}

// Async global->LDS 16B copy (CDNA5 GLOBAL_LOAD_ASYNC_TO_LDS_B128, ASYNCcnt).
// VDST operand = LDS byte address (low 32 bits of the generic shared pointer,
// per the flat-aperture mapping).
__device__ __forceinline__ void async_copy16(void* lds_dst, const void* gsrc) {
  uint32_t l = (uint32_t)(size_t)lds_dst;
  uint64_t g = (uint64_t)(size_t)gsrc;
  asm volatile("global_load_async_to_lds_b128 %0, %1, off"
               :: "v"(l), "v"(g) : "memory");
}
#define WAIT_ASYNC(n) asm volatile("s_wait_asynccnt " #n ::: "memory")

// Load a 16x32 bf16 WMMA operand fragment from a row-major tile with 16B chunks.
// Per CDNA5 ISA layout: lanes 0-15 hold K{kbase+0..7, kbase+16..23},
// lanes 16-31 hold K{kbase+8..15, kbase+24..31}; row chosen by caller via rowptr.
__device__ __forceinline__ v16bf ld_frag(const bf16* rowptr, int kbase, int lane) {
  FragU u;
  int ko = kbase + ((lane & 16) ? 8 : 0);
  u.q[0] = *(const uint4*)(rowptr + ko);
  u.q[1] = *(const uint4*)(rowptr + ko + 16);
  return u.v;
}

// ---------------------------------------------------------------------------
// fp32 -> bf16 conversion
// ---------------------------------------------------------------------------
__global__ __launch_bounds__(256) void cvt_bf16_kernel(const float* __restrict__ src,
                                                       bf16* __restrict__ dst, int n) {
  int i = blockIdx.x * 256 + threadIdx.x;
  if (i < n) dst[i] = (bf16)src[i];
}

// ---------------------------------------------------------------------------
// C[M,N] (f32) = A[M,K] (bf16, row major) @ W[N,K]^T (bf16, row major)
// Block tile 128x128, K-step 32, 256 threads = 8 waves (4 along M, 2 along N).
// Double-buffered LDS fed by async global->LDS copies (ASYNCcnt), WMMA compute.
// ---------------------------------------------------------------------------
__global__ __launch_bounds__(256) void gemm_bf16_wmma(
    const bf16* __restrict__ A, const bf16* __restrict__ W,
    float* __restrict__ C, int M, int N, int K)
{
  __shared__ __align__(16) bf16 As[2][128][32];
  __shared__ __align__(16) bf16 Bs[2][128][32];
  const int tid  = threadIdx.x;
  const int lane = tid & 31;
  const int wave = tid >> 5;
  const int wm = wave & 3;   // wave row (0..3) -> 32 M-rows each
  const int wn = wave >> 2;  // wave col (0..1) -> 64 N-cols each
  const int m0 = blockIdx.y * 128;
  const int n0 = blockIdx.x * 128;
  const int lrow = lane & 15;

  v8f acc[2][4];
#pragma unroll
  for (int mt = 0; mt < 2; mt++)
#pragma unroll
    for (int nt = 0; nt < 4; nt++) acc[mt][nt] = zero8();

  // Issue one K-stage (32 cols) of A+B tiles: 4 async b128 per thread.
  auto issue_stage = [&](int ks, int buf) {
    const int k0 = ks * 32;
    for (int c = tid; c < 512; c += 256) {
      int r = c >> 2, qc = (c & 3) * 8;
      async_copy16(&As[buf][r][qc], A + (size_t)(m0 + r) * K + k0 + qc);
      async_copy16(&Bs[buf][r][qc], W + (size_t)(n0 + r) * K + k0 + qc);
    }
  };

  const int nst = K / 32;
  issue_stage(0, 0);
  for (int ks = 0; ks < nst; ks++) {
    const int buf = ks & 1;
    if (ks + 1 < nst) {
      issue_stage(ks + 1, buf ^ 1);
      WAIT_ASYNC(4);   // stage ks landed; stage ks+1 may remain in flight
    } else {
      WAIT_ASYNC(0);
    }
    __syncthreads();

    v16bf af[2], bf_[4];
#pragma unroll
    for (int mt = 0; mt < 2; mt++)
      af[mt] = ld_frag(&As[buf][wm * 32 + mt * 16 + lrow][0], 0, lane);
#pragma unroll
    for (int nt = 0; nt < 4; nt++)
      bf_[nt] = ld_frag(&Bs[buf][wn * 64 + nt * 16 + lrow][0], 0, lane);

#pragma unroll
    for (int mt = 0; mt < 2; mt++)
#pragma unroll
      for (int nt = 0; nt < 4; nt++)
        acc[mt][nt] = __builtin_amdgcn_wmma_f32_16x16x32_bf16(
            false, af[mt], false, bf_[nt], (short)0, acc[mt][nt], false, false);
    __syncthreads();   // all waves done reading buf before it is re-filled
  }

  // Store (C layout: VGPR v -> row v + 8*(lane>=16), col = lane&15 within tile)
  const int rhi = (lane >> 4) << 3;
#pragma unroll
  for (int mt = 0; mt < 2; mt++)
#pragma unroll
    for (int nt = 0; nt < 4; nt++)
#pragma unroll
      for (int v = 0; v < 8; v++) {
        int row = m0 + wm * 32 + mt * 16 + v + rhi;
        int col = n0 + wn * 64 + nt * 16 + lrow;
        C[(size_t)row * N + col] = acc[mt][nt][v];
      }
}

// ---------------------------------------------------------------------------
// RoPE + RMSNorm, fused. One block per (b, s, head). 128 threads = D.
// Input: proj [B*S, Hx*D] f32. Output: [B, Hx, S, D] bf16.
// ---------------------------------------------------------------------------
__global__ __launch_bounds__(128) void rope_rms_kernel(
    const float* __restrict__ proj, const float* __restrict__ cosb,
    const float* __restrict__ sinb, bf16* __restrict__ out, int Hx)
{
  int idx = blockIdx.x;
  int h = idx % Hx;
  int s = (idx / Hx) % S_;
  int b = idx / (Hx * S_);
  int d = threadIdx.x;
  const float* xr = proj + (size_t)(b * S_ + s) * (Hx * D_) + h * D_;

  float y;
  if (d < 64) {
    float c = cosb[s * 64 + d], sn = sinb[s * 64 + d];
    y = xr[d] * c + xr[d + 64] * sn;
  } else {
    float c = cosb[s * 64 + d - 64], sn = sinb[s * 64 + d - 64];
    y = -xr[d - 64] * sn + xr[d] * c;
  }

  __shared__ float red[128];
  red[d] = y * y;
  __syncthreads();
  for (int off = 64; off > 0; off >>= 1) {
    if (d < off) red[d] += red[d + off];
    __syncthreads();
  }
  float scale = rsqrtf(red[0] * (1.0f / 128.0f) + 1.1920929e-7f);
  out[(((size_t)(b * Hx + h)) * S_ + s) * D_ + d] = (bf16)(y * scale);
}

// ---------------------------------------------------------------------------
// V: f32 [B*S, HKV*D] -> bf16 transposed [B, HKV, D, S] (so the PV WMMA
// B-operand column (fixed d, K=j contiguous) is row-contiguous in memory).
// ---------------------------------------------------------------------------
__global__ __launch_bounds__(256) void v_cvt_transpose_kernel(
    const float* __restrict__ vproj, bf16* __restrict__ vt)
{
  size_t i = (size_t)blockIdx.x * 256 + threadIdx.x;
  if (i >= (size_t)B_ * S_ * HKV_ * D_) return;
  int d = (int)(i % D_);
  int h = (int)((i / D_) % HKV_);
  size_t bs = i / (D_ * HKV_);
  int s = (int)(bs % S_);
  int b = (int)(bs / S_);
  vt[(((size_t)(b * HKV_ + h)) * D_ + d) * S_ + s] = (bf16)vproj[i];
}

// ---------------------------------------------------------------------------
// Causal flash attention. Grid: (S/64, B*H). 128 threads = 4 waves, each wave
// owns 16 q-rows. K/V streamed in 64-row tiles through double-buffered LDS via
// async global->LDS copies; QK^T and PV via WMMA; online softmax in registers.
// ---------------------------------------------------------------------------
__global__ __launch_bounds__(128) void flash_attn_kernel(
    const bf16* __restrict__ qn, const bf16* __restrict__ kn,
    const bf16* __restrict__ vt, bf16* __restrict__ attn)
{
  __shared__ __align__(16) bf16 Ks[2][64][128];   // 32 KB: K tiles [j][d]
  __shared__ __align__(16) bf16 Vs[2][128][64];   // 32 KB: V tiles [d][j]
  __shared__ __align__(16) bf16 Ps[4][16][64];    //  8 KB: per-wave P staging

  const int tid = threadIdx.x, lane = tid & 31, wave = tid >> 5;
  const int bh = blockIdx.y, h = bh % H_, b = bh / H_, hk = h / (H_ / HKV_);
  const int q0 = blockIdx.x * 64;
  const int lrow = lane & 15;
  const int lhi = (lane >> 4) << 3;  // 0 or 8

  const bf16* qb = qn + ((size_t)(b * H_ + h)) * S_ * D_;
  const bf16* kb = kn + ((size_t)(b * HKV_ + hk)) * S_ * D_;
  const bf16* vb = vt + ((size_t)(b * HKV_ + hk)) * D_ * S_;

  // Resident Q fragments: 16 rows x 128 (4 K-steps of 32)
  v16bf qf[4];
  {
    const bf16* qr = qb + (size_t)(q0 + wave * 16 + lrow) * D_;
#pragma unroll
    for (int t = 0; t < 4; t++) qf[t] = ld_frag(qr, 32 * t, lane);
  }

  float m_i[8], l_i[8];
  v8f acc[8];
#pragma unroll
  for (int v = 0; v < 8; v++) { m_i[v] = -1e30f; l_i[v] = 0.f; acc[v] = zero8(); }

  const float sc = 0.08838834764831845f;  // 1/sqrt(128)

  // Issue K+V tiles for k-block jt: 16 async b128 per thread.
  auto issue_kv = [&](int jt, int buf) {
    const int j0 = jt * 64;
    for (int c = tid; c < 1024; c += 128) {
      int r = c >> 4, qc = (c & 15) * 8;
      async_copy16(&Ks[buf][r][qc], kb + (size_t)(j0 + r) * D_ + qc);
    }
    for (int c = tid; c < 1024; c += 128) {
      int r = c >> 3, qc = (c & 7) * 8;
      async_copy16(&Vs[buf][r][qc], vb + (size_t)r * S_ + j0 + qc);
    }
  };

  const int ntiles = q0 / 64 + 1;
  issue_kv(0, 0);
  for (int jt = 0; jt < ntiles; jt++) {
    const int j0 = jt * 64;
    const int buf = jt & 1;
    if (jt + 1 < ntiles) {
      issue_kv(jt + 1, buf ^ 1);
      WAIT_ASYNC(16);  // tile jt landed; tile jt+1 may remain in flight
    } else {
      WAIT_ASYNC(0);
    }
    __syncthreads();

    // S = Q @ K^T  (wave: 16x64 = 4 n-tiles, K-loop 4x32)
    v8f sf[4];
#pragma unroll
    for (int nt = 0; nt < 4; nt++) {
      sf[nt] = zero8();
#pragma unroll
      for (int t = 0; t < 4; t++) {
        v16bf bk = ld_frag(&Ks[buf][nt * 16 + lrow][0], 32 * t, lane);
        sf[nt] = __builtin_amdgcn_wmma_f32_16x16x32_bf16(
            false, qf[t], false, bk, (short)0, sf[nt], false, false);
      }
    }

    // Scale + causal mask + online softmax (half-wave shuffles per row)
    float pv[4][8];
#pragma unroll
    for (int v = 0; v < 8; v++) {
      int rg = q0 + wave * 16 + v + lhi;
      float mx = -1e30f;
#pragma unroll
      for (int nt = 0; nt < 4; nt++) {
        int cg = j0 + nt * 16 + lrow;
        float val = (cg <= rg) ? sf[nt][v] * sc : -1e30f;
        pv[nt][v] = val;
        mx = fmaxf(mx, val);
      }
#pragma unroll
      for (int off = 1; off < 16; off <<= 1) mx = fmaxf(mx, __shfl_xor(mx, off, 32));

      float mnew = fmaxf(m_i[v], mx);
      float alpha = __expf(m_i[v] - mnew);
      float rs = 0.f;
#pragma unroll
      for (int nt = 0; nt < 4; nt++) {
        float p = __expf(pv[nt][v] - mnew);
        pv[nt][v] = p;
        rs += p;
      }
#pragma unroll
      for (int off = 1; off < 16; off <<= 1) rs += __shfl_xor(rs, off, 32);
      l_i[v] = l_i[v] * alpha + rs;
      m_i[v] = mnew;
#pragma unroll
      for (int dt = 0; dt < 8; dt++) acc[dt][v] *= alpha;
    }

    // P (C-layout) -> LDS -> A-layout fragments (wave-private; DS in-order)
#pragma unroll
    for (int nt = 0; nt < 4; nt++)
#pragma unroll
      for (int v = 0; v < 8; v++)
        Ps[wave][v + lhi][nt * 16 + lrow] = (bf16)pv[nt][v];
    asm volatile("" ::: "memory");

    // O += P @ V  (K=64 -> 2 K-steps of 32, 8 d-tiles of 16)
#pragma unroll
    for (int tt = 0; tt < 2; tt++) {
      v16bf pa = ld_frag(&Ps[wave][lrow][0], 32 * tt, lane);
#pragma unroll
      for (int dt = 0; dt < 8; dt++) {
        v16bf vf = ld_frag(&Vs[buf][dt * 16 + lrow][0], 32 * tt, lane);
        acc[dt] = __builtin_amdgcn_wmma_f32_16x16x32_bf16(
            false, pa, false, vf, (short)0, acc[dt], false, false);
      }
    }
    __syncthreads();  // done reading buf before it is re-filled
  }

  // Epilogue: normalize and write bf16 [B*S, H*D]
#pragma unroll
  for (int v = 0; v < 8; v++) {
    int s = q0 + wave * 16 + v + lhi;
    float inv = 1.f / l_i[v];
#pragma unroll
    for (int dt = 0; dt < 8; dt++) {
      int d = dt * 16 + lrow;
      attn[(size_t)(b * S_ + s) * (H_ * D_) + h * D_ + d] = (bf16)(acc[dt][v] * inv);
    }
  }
}

// ---------------------------------------------------------------------------
// Host-side launcher
// ---------------------------------------------------------------------------
extern "C" void kernel_launch(void* const* d_in, const int* in_sizes, int n_in,
                              void* d_out, int out_size, void* d_ws, size_t ws_size,
                              hipStream_t stream) {
  (void)in_sizes; (void)n_in; (void)out_size; (void)ws_size;
  const float* x    = (const float*)d_in[0];
  const float* cosb = (const float*)d_in[1];
  const float* sinb = (const float*)d_in[2];
  const float* Wq   = (const float*)d_in[3];
  const float* Wk   = (const float*)d_in[4];
  const float* Wv   = (const float*)d_in[5];
  const float* Wo   = (const float*)d_in[6];
  float* out = (float*)d_out;

  const size_t MS = (size_t)B_ * S_;           // 4096 token rows
  const size_t NQ = (size_t)H_ * D_;           // 2048
  const size_t NKV = (size_t)HKV_ * D_;        // 512

  char* ws = (char*)d_ws;
  size_t off = 0;
  auto alloc = [&](size_t bytes) -> void* {
    void* p = ws + off;
    off += (bytes + 255) & ~(size_t)255;
    return p;
  };
  bf16*  xb   = (bf16*)alloc(MS * HID_ * sizeof(bf16));
  bf16*  Wqb  = (bf16*)alloc(NQ * HID_ * sizeof(bf16));
  bf16*  Wkb  = (bf16*)alloc(NKV * HID_ * sizeof(bf16));
  bf16*  Wvb  = (bf16*)alloc(NKV * HID_ * sizeof(bf16));
  bf16*  Wob  = (bf16*)alloc((size_t)HID_ * NQ * sizeof(bf16));
  float* qp   = (float*)alloc(MS * NQ * sizeof(float));
  float* kp   = (float*)alloc(MS * NKV * sizeof(float));
  float* vp   = (float*)alloc(MS * NKV * sizeof(float));
  bf16*  qnb  = (bf16*)alloc(MS * NQ * sizeof(bf16));
  bf16*  knb  = (bf16*)alloc(MS * NKV * sizeof(bf16));
  bf16*  vtb  = (bf16*)alloc(MS * NKV * sizeof(bf16));
  bf16*  attn = (bf16*)alloc(MS * NQ * sizeof(bf16));

  auto cvt = [&](const float* s, bf16* d, size_t n) {
    cvt_bf16_kernel<<<dim3((unsigned)((n + 255) / 256)), dim3(256), 0, stream>>>(s, d, (int)n);
  };
  // 1) fp32 -> bf16
  cvt(x,  xb,  MS * HID_);
  cvt(Wq, Wqb, NQ * HID_);
  cvt(Wk, Wkb, NKV * HID_);
  cvt(Wv, Wvb, NKV * HID_);
  cvt(Wo, Wob, (size_t)HID_ * NQ);

  // 2) projections (C = A @ W^T)
  gemm_bf16_wmma<<<dim3((unsigned)(NQ / 128),  (unsigned)(MS / 128)), dim3(256), 0, stream>>>(
      xb, Wqb, qp, (int)MS, (int)NQ, HID_);
  gemm_bf16_wmma<<<dim3((unsigned)(NKV / 128), (unsigned)(MS / 128)), dim3(256), 0, stream>>>(
      xb, Wkb, kp, (int)MS, (int)NKV, HID_);
  gemm_bf16_wmma<<<dim3((unsigned)(NKV / 128), (unsigned)(MS / 128)), dim3(256), 0, stream>>>(
      xb, Wvb, vp, (int)MS, (int)NKV, HID_);

  // 3) RoPE + RMSNorm (q, k) and V convert/transpose
  rope_rms_kernel<<<dim3((unsigned)(MS * H_)),   dim3(128), 0, stream>>>(qp, cosb, sinb, qnb, H_);
  rope_rms_kernel<<<dim3((unsigned)(MS * HKV_)), dim3(128), 0, stream>>>(kp, cosb, sinb, knb, HKV_);
  v_cvt_transpose_kernel<<<dim3((unsigned)((MS * NKV + 255) / 256)), dim3(256), 0, stream>>>(vp, vtb);

  // 4) causal flash attention
  flash_attn_kernel<<<dim3(S_ / 64, B_ * H_), dim3(128), 0, stream>>>(qnb, knb, vtb, attn);

  // 5) output projection -> d_out (f32)
  gemm_bf16_wmma<<<dim3((unsigned)(HID_ / 128), (unsigned)(MS / 128)), dim3(256), 0, stream>>>(
      attn, Wob, out, (int)MS, HID_, (int)NQ);
}
